// ScaledDotProductAttention_10273561772131
// MI455X (gfx1250) — matrix operand order repaired and checked by reading, not compiled, not run on hardware
//
#include <hip/hip_runtime.h>
#include <hip/hip_bf16.h>
#include <stdint.h>

typedef __attribute__((ext_vector_type(16))) _Float16 v16h;
typedef __attribute__((ext_vector_type(8)))  float    v8f;

#define B_ 8
#define H_ 8
#define N_ 1024
#define D_ 64
#define INV_TEMP 0.125f
#define LDS_STRIDE 68          // padded row stride in floats (272 B, 16B-aligned, kills bank conflicts)
#define BUF_FLOATS (32 * LDS_STRIDE)
#define BUF_BYTES  (BUF_FLOATS * 4)

// ---- CDNA5 async global->LDS copy (ASYNCcnt-tracked), per CDNA5 ISA
__device__ __forceinline__ void async_copy_b128(uint32_t lds_addr, const void* gptr) {
  uint64_t ga = (uint64_t)(uintptr_t)gptr;
  asm volatile("global_load_async_to_lds_b128 %0, %1, off" :: "v"(lds_addr), "v"(ga) : "memory");
}
__device__ __forceinline__ void wait_async_le(int n) {  // n is block-uniform
  if (n == 0)      asm volatile("s_wait_asynccnt 0" ::: "memory");
  else if (n == 1) asm volatile("s_wait_asynccnt 1" ::: "memory");
  else             asm volatile("s_wait_asynccnt 2" ::: "memory");
}
__device__ __forceinline__ void fence_stores() {
  asm volatile("s_wait_storecnt 0" ::: "memory");
}

__global__ __launch_bounds__(256, 1)
void sdpa_wmma_kernel(const float* __restrict__ q, const float* __restrict__ k,
                      const float* __restrict__ v, const float* __restrict__ cost,
                      const int*   __restrict__ mask,
                      float* __restrict__ out, float* __restrict__ attn)
{
  __shared__ alignas(16) float kv[2 * BUF_FLOATS];   // double-buffered K/V staging (17.4 KB)
  const uint32_t kvBase = (uint32_t)(uintptr_t)&kv[0];  // low 32 bits of flat addr == LDS offset

  const int tid  = threadIdx.x;
  const int lane = tid & 31;
  const int wave = tid >> 5;
  const int m16  = lane & 15;   // row (A/C) or col (B) within 16
  const int hi   = lane >> 4;   // half-wave select

  const int bh = blockIdx.x >> 3;      // (b*H + h)
  const int rb = blockIdx.x & 7;       // 128-row block
  const int b  = bh >> 3;              // bh / H_
  const int i0 = rb * 128 + wave * 16; // this wave's 16-row tile

  const float* qBH = q + (size_t)bh * N_ * D_;
  const float* kBH = k + (size_t)bh * N_ * D_;
  const float* vBH = v + (size_t)bh * N_ * D_;
  float* outBH  = out  + (size_t)bh * N_ * D_;
  float* attnBH = attn + (size_t)bh * N_ * N_;
  const int*   maskB = mask + (size_t)b * N_ * N_;
  const float* costB = cost + (size_t)b * N_ * N_;

  // ---------------- preload Q A-fragments (two K=32 chunks), scaled by 1/T
  // A operand (this code's contiguous k order): lane holds row m16; half h -> K = hi*16 + h
  v16h aq[2];
  {
    const float* qrow = qBH + (size_t)(i0 + m16) * D_;
    #pragma unroll
    for (int c = 0; c < 2; ++c) {
      #pragma unroll
      for (int t = 0; t < 8; ++t) {
        const int h    = 2 * t;
        const int koff = hi * 16 + h;
        float2 f = *(const float2*)(qrow + c * 32 + koff);
        aq[c][h]     = (_Float16)(f.x * INV_TEMP);
        aq[c][h + 1] = (_Float16)(f.y * INV_TEMP);
      }
    }
  }

  // ---------------- Phase 1: raw scores S = (Q/T) K^T  -> global attn buffer
  // Software-pipelined: async-load K tile j+1 while computing tile j.
  {
    const int row  = tid >> 4;   // 0..15 : K-tile row this thread copies
    const int part = tid & 15;   // 16B chunk within the 256B row
    const uint32_t ldsDst = kvBase + (uint32_t)(row * (LDS_STRIDE * 4) + part * 16);
    const char* gsrc = (const char*)kBH + (size_t)row * (D_ * 4) + part * 16;

    async_copy_b128(ldsDst, gsrc);                      // prologue: tile 0 -> buf 0
    for (int j = 0; j < N_ / 16; ++j) {
      __syncthreads();                                  // buf[(j+1)&1] free (readers of j-1 done)
      const bool more = (j + 1) < (N_ / 16);
      if (more)
        async_copy_b128(ldsDst + ((j + 1) & 1) * BUF_BYTES,
                        gsrc + (size_t)(j + 1) * 16 * D_ * 4);
      wait_async_le(more ? 1 : 0);                      // my tile-j chunk resident
      __syncthreads();                                  // everyone's tile-j chunks resident

      const float* buf = &kv[(j & 1) * BUF_FLOATS];
      v8f acc = {};
      #pragma unroll
      for (int c = 0; c < 2; ++c) {
        // B operand (same contiguous k order as A): lane holds col n=m16; half h -> k = hi*16 + h
        // B[k][n] = K[j*16+n][c*32 + k]
        v16h bf;
        const float* krow = buf + m16 * LDS_STRIDE + c * 32 + hi * 16;
        #pragma unroll
        for (int t = 0; t < 8; ++t) {
          float2 f = *(const float2*)(krow + 2 * t);
          bf[2 * t]     = (_Float16)f.x;
          bf[2 * t + 1] = (_Float16)f.y;
        }
        acc = __builtin_amdgcn_wmma_f32_16x16x32_f16(false, aq[c], false, bf,
                                                     (short)0, acc, false, false);
      }
      // C/D layout: VGPR r -> row (r + 8*hi), col = j*16 + m16
      float* dst = attnBH + (size_t)(i0 + hi * 8) * N_ + j * 16 + m16;
      #pragma unroll
      for (int r = 0; r < 8; ++r)
        dst[(size_t)r * N_] = acc[r];
    }
  }
  fence_stores();  // scores visible for cross-lane re-read (same CU/L0)

  // ---------------- Phase 2: masked softmax + post mask*cost (wave-local, coalesced)
  for (int r = 0; r < 16; ++r) {
    const int rowG = i0 + r;
    float*       sRow = attnBH + (size_t)rowG * N_;
    const int*   mRow = maskB  + (size_t)rowG * N_;
    const float* cRow = costB  + (size_t)rowG * N_;

    float mx = -3.0e38f;
    for (int t = 0; t < 32; ++t) {
      const int c = t * 32 + lane;
      float s = sRow[c];
      s = (mRow[c] == 0) ? -1.0e9f : s;   // pre-softmax mask
      sRow[c] = s;
      mx = fmaxf(mx, s);
    }
    #pragma unroll
    for (int o = 16; o > 0; o >>= 1) mx = fmaxf(mx, __shfl_xor(mx, o, 32));

    float sum = 0.0f;
    for (int t = 0; t < 32; ++t) {
      const int c = t * 32 + lane;
      const float e = __expf(sRow[c] - mx);  // masked entries underflow to exact 0
      sRow[c] = e;
      sum += e;
    }
    #pragma unroll
    for (int o = 16; o > 0; o >>= 1) sum += __shfl_xor(sum, o, 32);

    // fully-masked row (mx == -1e9): reference post-mask zeroes it -> inv = 0
    const float inv = (mx <= -1.0e8f) ? 0.0f : (1.0f / sum);
    for (int t = 0; t < 32; ++t) {
      const int c = t * 32 + lane;
      sRow[c] = sRow[c] * inv * cRow[c];     // post-softmax mask*cost
    }
  }
  fence_stores();  // final attn visible for WMMA A-fragment re-read

  // ---------------- Phase 3: O = attn @ V  (V tiles async-staged, double-buffered)
  {
    v8f acc[4] = {{}, {}, {}, {}};
    // copy plan: 32 rows x 256B = 512 x 16B chunks / 256 threads -> 2 chunks each
    const int c0row  = tid >> 4,          c0part = tid & 15;
    const int c1row  = (tid + 256) >> 4,  c1part = tid & 15;
    const uint32_t ldsDst0 = kvBase + (uint32_t)(c0row * (LDS_STRIDE * 4) + c0part * 16);
    const uint32_t ldsDst1 = kvBase + (uint32_t)(c1row * (LDS_STRIDE * 4) + c1part * 16);
    const char* gs0 = (const char*)vBH + (size_t)c0row * (D_ * 4) + c0part * 16;
    const char* gs1 = (const char*)vBH + (size_t)c1row * (D_ * 4) + c1part * 16;

    async_copy_b128(ldsDst0, gs0);                      // prologue: k-tile 0 -> buf 0
    async_copy_b128(ldsDst1, gs1);
    for (int kt = 0; kt < 32; ++kt) {
      __syncthreads();                                  // buf[(kt+1)&1] free
      const bool more = (kt + 1) < 32;
      if (more) {
        const uint32_t bb = ((kt + 1) & 1) * BUF_BYTES;
        const size_t   gb = (size_t)(kt + 1) * 32 * D_ * 4;
        async_copy_b128(ldsDst0 + bb, gs0 + gb);
        async_copy_b128(ldsDst1 + bb, gs1 + gb);
      }
      wait_async_le(more ? 2 : 0);                      // my k-tile chunks resident
      __syncthreads();                                  // everyone's chunks resident

      const float* buf = &kv[(kt & 1) * BUF_FLOATS];

      // A fragment from final attn (global, L2-hot): A[m][k], k = Koff(h)
      v16h af;
      {
        const float* arow = attnBH + (size_t)(i0 + m16) * N_ + kt * 32;
        #pragma unroll
        for (int t = 0; t < 8; ++t) {
          const int h    = 2 * t;
          const int koff = hi * 16 + h;
          float2 f = *(const float2*)(arow + koff);
          af[h]     = (_Float16)f.x;
          af[h + 1] = (_Float16)f.y;
        }
      }
      // B fragments from LDS V tile: B[k][n] = V[kt*32 + k][dt*16 + n], k = hi*16 + h
      #pragma unroll
      for (int dt = 0; dt < 4; ++dt) {
        v16h bf;
        #pragma unroll
        for (int h = 0; h < 16; ++h)
          bf[h] = (_Float16)buf[(hi * 16 + h) * LDS_STRIDE + dt * 16 + m16];
        acc[dt] = __builtin_amdgcn_wmma_f32_16x16x32_f16(false, af, false, bf,
                                                         (short)0, acc[dt], false, false);
      }
    }
    // write output tile: row (r + 8*hi), col dt*16 + m16 (coalesced)
    #pragma unroll
    for (int dt = 0; dt < 4; ++dt) {
      float* dst = outBH + (size_t)(i0 + hi * 8) * D_ + dt * 16 + m16;
      #pragma unroll
      for (int r = 0; r < 8; ++r)
        dst[(size_t)r * D_] = acc[dt][r];
    }
  }
}

extern "C" void kernel_launch(void* const* d_in, const int* in_sizes, int n_in,
                              void* d_out, int out_size, void* d_ws, size_t ws_size,
                              hipStream_t stream) {
  const float* q    = (const float*)d_in[0];
  const float* k    = (const float*)d_in[1];
  const float* v    = (const float*)d_in[2];
  const float* cost = (const float*)d_in[3];
  const int*   mask = (const int*)d_in[4];
  float* out  = (float*)d_out;
  float* attn = out + (size_t)B_ * H_ * N_ * D_;   // tuple: (output, attn) flat-concat

  dim3 grid(B_ * H_ * (N_ / 128));
  dim3 block(256);
  hipLaunchKernelGGL(sdpa_wmma_kernel, grid, block, 0, stream,
                     q, k, v, cost, mask, out, attn);
}
